// SelfAttention_86071144612365
// MI455X (gfx1250) — matrix operand order repaired and checked by reading, not compiled, not run on hardware
//
#include <hip/hip_runtime.h>

// ---------------------------------------------------------------------------
// Self-attention block (GroupNorm -> QKV -> flash-attention -> out-proj + res)
// B=8, S=4096, C=256, G=32.  GEMMs via v_wmma_f32_16x16x32_f16; K/V tiles
// staged into LDS with global_load_async_to_lds_b128 (ASYNCcnt).
// ---------------------------------------------------------------------------

typedef __attribute__((ext_vector_type(16))) _Float16 v16h;
typedef __attribute__((ext_vector_type(8)))  _Float16 v8h;
typedef __attribute__((ext_vector_type(8)))  float    v8f;

constexpr int kB = 8, kS = 4096, kC = 256;

// workspace layout (bytes)
constexpr size_t OFF_HF   = 4096;                         // hf16 / O16 (reused)
constexpr size_t HF_BYTES = (size_t)kB * kS * kC * 2;     // 16.78 MB
constexpr size_t OFF_Q    = OFF_HF + HF_BYTES;
constexpr size_t OFF_K    = OFF_Q  + HF_BYTES;
constexpr size_t OFF_V    = OFF_K  + HF_BYTES;            // V stored transposed (C x S)
constexpr size_t OFF_W    = OFF_V  + HF_BYTES;            // 4 packed 256x256 f16 weights

__device__ __forceinline__ v8f vzero() {
  v8f z;
#pragma unroll
  for (int j = 0; j < 8; ++j) z[j] = 0.0f;
  return z;
}

__device__ __forceinline__ v8f wmma_f16(v16h a, v16h b, v8f c) {
  return __builtin_amdgcn_wmma_f32_16x16x32_f16(false, a, false, b, (short)0, c,
                                                false, false);
}

// A fragment 16x32 f16: lane h=l>>4 holds K = {h*16..h*16+15}
__device__ __forceinline__ v16h ld_afrag(const _Float16* base, int h) {
  union { v16h v; v8h p[2]; } u;
  u.p[0] = *(const v8h*)(base + h * 16);
  u.p[1] = *(const v8h*)(base + h * 16 + 8);
  return u.v;
}
// B fragment 32x16 f16: lane column contiguous in K (K = h*16 + i)
__device__ __forceinline__ v16h ld_bfrag(const _Float16* base, int h) {
  return *(const v16h*)(base + h * 16);
}

// async copy 16B/lane: LDS[lds_byte_off] = *gptr  (tracked by ASYNCcnt)
__device__ __forceinline__ void async_g2l_b128(unsigned lds_byte_off, const void* gptr) {
  asm volatile("global_load_async_to_lds_b128 %0, %1, off"
               :
               : "v"(lds_byte_off), "v"(gptr)
               : "memory");
}
__device__ __forceinline__ void wait_async0() {
  asm volatile("s_wait_asynccnt 0x0" ::: "memory");
}
__device__ __forceinline__ unsigned lds_off(const void* p) {
  return (unsigned)(uintptr_t)p;   // generic shared addr: low 32 bits = LDS offset
}
__device__ __forceinline__ void wave_fence() {
  __builtin_amdgcn_fence(__ATOMIC_ACQ_REL, "wavefront");
}

__device__ __forceinline__ float rmax16(float v) {
  v = fmaxf(v, __shfl_xor(v, 1, 32));
  v = fmaxf(v, __shfl_xor(v, 2, 32));
  v = fmaxf(v, __shfl_xor(v, 4, 32));
  v = fmaxf(v, __shfl_xor(v, 8, 32));
  return v;
}
__device__ __forceinline__ float rsum16(float v) {
  v += __shfl_xor(v, 1, 32);
  v += __shfl_xor(v, 2, 32);
  v += __shfl_xor(v, 4, 32);
  v += __shfl_xor(v, 8, 32);
  return v;
}

// ---------------- GroupNorm statistics: one block per (batch, group) --------
__global__ __launch_bounds__(256) void sa_gn_stats(const float* __restrict__ x,
                                                   float* __restrict__ stats) {
  const int g = blockIdx.x, b = blockIdx.y, t = threadIdx.x;
  const float* base = x + (size_t)b * kS * kC + g * 8;
  float s = 0.f, ss = 0.f;
  for (int row = t; row < kS; row += 256) {
    const float4* p = (const float4*)(base + (size_t)row * kC);
    float4 a = p[0], c = p[1];
    s  += a.x + a.y + a.z + a.w + c.x + c.y + c.z + c.w;
    ss += a.x*a.x + a.y*a.y + a.z*a.z + a.w*a.w +
          c.x*c.x + c.y*c.y + c.z*c.z + c.w*c.w;
  }
  __shared__ float sh0[256], sh1[256];
  sh0[t] = s; sh1[t] = ss;
  __syncthreads();
  for (int o = 128; o > 0; o >>= 1) {
    if (t < o) { sh0[t] += sh0[t + o]; sh1[t] += sh1[t + o]; }
    __syncthreads();
  }
  if (t == 0) {
    const float inv_n = 1.0f / (float)(kS * 8);
    float mean = sh0[0] * inv_n;
    float var  = sh1[0] * inv_n - mean * mean;
    stats[(b * 32 + g) * 2 + 0] = mean;
    stats[(b * 32 + g) * 2 + 1] = rsqrtf(var + 1e-6f);
  }
}

// ---------------- Apply GN, emit f16 activations ---------------------------
__global__ __launch_bounds__(256) void sa_gn_apply(const float* __restrict__ x,
                                                   const float* __restrict__ scale,
                                                   const float* __restrict__ bias,
                                                   const float* __restrict__ stats,
                                                   _Float16* __restrict__ hf) {
  size_t id = (size_t)blockIdx.x * 256 + threadIdx.x;   // one thread = 8 channels
  int    c8 = (int)(id & 31);                           // == group index (C/G == 8)
  size_t bs = id >> 5;                                  // 0 .. B*S-1
  int    b  = (int)(bs >> 12);
  float mean = stats[(b * 32 + c8) * 2 + 0];
  float rstd = stats[(b * 32 + c8) * 2 + 1];
  const float4* px = (const float4*)(x + bs * kC + c8 * 8);
  float4 a = px[0], d = px[1];
  float v[8] = {a.x, a.y, a.z, a.w, d.x, d.y, d.z, d.w};
  v8h o;
#pragma unroll
  for (int j = 0; j < 8; ++j)
    o[j] = (_Float16)((v[j] - mean) * rstd * scale[c8 * 8 + j] + bias[c8 * 8 + j]);
  *(v8h*)(hf + bs * kC + c8 * 8) = o;
}

// ---------------- Pack fp32 weights into WMMA B-fragment order -------------
__global__ __launch_bounds__(256) void sa_pack_w(const float* __restrict__ wq,
                                                 const float* __restrict__ wk,
                                                 const float* __restrict__ wv,
                                                 const float* __restrict__ wo,
                                                 _Float16* __restrict__ wpack) {
  int e = blockIdx.x * 256 + threadIdx.x;               // 0..65535
  int m = blockIdx.y;
  const float* w = (m == 0) ? wq : (m == 1) ? wk : (m == 2) ? wv : wo;
  int i    = e & 15;
  int lane = (e >> 4) & 31;
  int kt   = (e >> 9) & 7;
  int nt   = e >> 12;
  int h    = lane >> 4;
  int n    = nt * 16 + (lane & 15);
  int k    = kt * 32 + h * 16 + i;                      // B-frag K map
  wpack[(size_t)m * 65536 + e] = (_Float16)w[k * kC + n];
}

// ---------------- QKV projection: wave = 16 rows x 256 cols ----------------
__global__ __launch_bounds__(256, 1) void sa_qkv(const _Float16* __restrict__ hf,
                                                 const _Float16* __restrict__ wpack,
                                                 const float* __restrict__ bq,
                                                 const float* __restrict__ bk,
                                                 const float* __restrict__ bv,
                                                 _Float16* __restrict__ Q,
                                                 _Float16* __restrict__ K,
                                                 _Float16* __restrict__ Vt) {
  const int lane = threadIdx.x & 31, wave = threadIdx.x >> 5;
  const int b = blockIdx.y, which = blockIdx.z;
  const int m0 = blockIdx.x * 128 + wave * 16;
  const int h = lane >> 4, r = lane & 15;
  const _Float16* wp = wpack + (size_t)which * 65536;
  const float* bias = (which == 0) ? bq : (which == 1) ? bk : bv;

  v8f acc[16];
#pragma unroll
  for (int nt = 0; nt < 16; ++nt) acc[nt] = vzero();

  const _Float16* arow = hf + ((size_t)(b * kS + m0 + r)) * kC;
  v16h a = ld_afrag(arow, h);                           // pipelined A fragment
#pragma unroll
  for (int kt = 0; kt < 8; ++kt) {
    v16h anext = a;
    if (kt < 7) anext = ld_afrag(arow + (kt + 1) * 32, h);
#pragma unroll
    for (int nt0 = 0; nt0 < 16; nt0 += 8) {             // 8 B-frags in flight
      v16h bb[8];
#pragma unroll
      for (int u = 0; u < 8; ++u)
        bb[u] = *(const v16h*)(wp + ((((nt0 + u) * 8 + kt) * 32 + lane) << 4));
#pragma unroll
      for (int u = 0; u < 8; ++u)
        acc[nt0 + u] = wmma_f16(a, bb[u], acc[nt0 + u]);
    }
    a = anext;
  }

#pragma unroll
  for (int nt = 0; nt < 16; ++nt) {
    int col = nt * 16 + r;
    float bval = bias[col];
    if (which == 2) {                       // V: store transposed (C x S)
      v8h o;
#pragma unroll
      for (int j = 0; j < 8; ++j) o[j] = (_Float16)(acc[nt][j] + bval);
      *(v8h*)(Vt + ((size_t)(b * kC + col)) * kS + m0 + h * 8) = o;
    } else {
      _Float16* dst = (which == 0) ? Q : K;
      float sc = (which == 0) ? 0.0625f : 1.0f;   // fold 1/sqrt(256) into Q
#pragma unroll
      for (int j = 0; j < 8; ++j) {
        int row = m0 + h * 8 + j;
        dst[((size_t)(b * kS + row)) * kC + col] = (_Float16)((acc[nt][j] + bval) * sc);
      }
    }
  }
}

// ---------------- Flash attention: block = 128 query rows ------------------
// K/V 32-key tiles staged once per block into LDS via async-to-LDS DMA.
__global__ __launch_bounds__(256, 1) void sa_attn(const _Float16* __restrict__ Q,
                                                  const _Float16* __restrict__ K,
                                                  const _Float16* __restrict__ Vt,
                                                  _Float16* __restrict__ O) {
  __shared__ _Float16 klds[32 * kC];        // 16 KB: keys t0..t0+31, row-major
  __shared__ _Float16 vlds[kC * 32];        // 16 KB: V^T cols, 32 keys each
  __shared__ _Float16 plds[8 * 512];        // 8 KB: per-wave P staging
  const int tid = threadIdx.x;
  const int lane = tid & 31, wave = tid >> 5;
  const int b = blockIdx.y;
  const int m0 = blockIdx.x * 128 + wave * 16;
  const int h = lane >> 4, r = lane & 15;
  _Float16* pl = plds + wave * 512;

  // preload Q fragments (reused across the whole key loop)
  v16h qa[8];
  const _Float16* qrow = Q + ((size_t)(b * kS + m0 + r)) * kC;
#pragma unroll
  for (int kt = 0; kt < 8; ++kt) qa[kt] = ld_afrag(qrow + kt * 32, h);

  v8f acc[16];
#pragma unroll
  for (int nt = 0; nt < 16; ++nt) acc[nt] = vzero();
  float mrow[8], lrow[8];
#pragma unroll
  for (int j = 0; j < 8; ++j) { mrow[j] = -1e30f; lrow[j] = 0.0f; }

  const _Float16* kb = K  + (size_t)b * kS * kC;
  const _Float16* vb = Vt + (size_t)b * kC * kS;
  const unsigned koff = lds_off(klds), voff = lds_off(vlds);

  for (int t0 = 0; t0 < kS; t0 += 32) {
    __syncthreads();                        // WAR: prior tile reads complete
    // ---- stage K tile (contiguous 16 KB) : 4 x 16B per thread ----
    const _Float16* gk = kb + (size_t)t0 * kC;
#pragma unroll
    for (int i = 0; i < 4; ++i) {
      int ci = tid + i * 256;               // 1024 chunks of 16B
      async_g2l_b128(koff + ci * 16, (const char*)gk + ci * 16);
    }
    // ---- stage V tile: per col 64B (strided source) ----
#pragma unroll
    for (int i = 0; i < 4; ++i) {
      int ci = tid + i * 256;               // 1024 chunks: col = ci>>2, part = ci&3
      int col = ci >> 2, part = ci & 3;
      async_g2l_b128(voff + (col * 32 + part * 8) * 2,
                     vb + (size_t)col * kS + t0 + part * 8);
    }
    // prefetch next tiles into L2 while async engine drains this one
    if (t0 + 32 < kS) {
      __builtin_prefetch(kb + (size_t)(t0 + 32) * kC + tid * 32, 0, 0);
      __builtin_prefetch(vb + (size_t)tid * kS + t0 + 32, 0, 0);
    }
    wait_async0();
    __syncthreads();                        // tile visible to all waves

    // ---- scores: 16x32 keys from LDS, contraction over C=256 ----
    v8f s0 = vzero(), s1 = vzero();
    const _Float16* k0p = klds + r * kC;
    const _Float16* k1p = klds + (16 + r) * kC;
#pragma unroll
    for (int kt = 0; kt < 8; kt += 2) {
      v16h b00 = ld_bfrag(k0p + kt * 32, h);
      v16h b01 = ld_bfrag(k1p + kt * 32, h);
      v16h b10 = ld_bfrag(k0p + kt * 32 + 32, h);
      v16h b11 = ld_bfrag(k1p + kt * 32 + 32, h);
      s0 = wmma_f16(qa[kt], b00, s0);
      s1 = wmma_f16(qa[kt], b01, s1);
      s0 = wmma_f16(qa[kt + 1], b10, s0);
      s1 = wmma_f16(qa[kt + 1], b11, s1);
    }

    // ---- online softmax (rows live across 16 lanes of a half-wave) ----
    float corr[8];
#pragma unroll
    for (int j = 0; j < 8; ++j) {
      float mx   = rmax16(fmaxf(s0[j], s1[j]));
      float mnew = fmaxf(mrow[j], mx);
      float c    = __expf(mrow[j] - mnew);
      float p0   = __expf(s0[j] - mnew);
      float p1   = __expf(s1[j] - mnew);
      s0[j] = p0; s1[j] = p1;
      lrow[j] = lrow[j] * c + rsum16(p0 + p1);
      mrow[j] = mnew;
      corr[j] = c;
    }
#pragma unroll
    for (int nt = 0; nt < 16; ++nt)
#pragma unroll
      for (int j = 0; j < 8; ++j) acc[nt][j] *= corr[j];

    // ---- transpose P (C/D layout -> A-frag layout); per-wave LDS region ----
    wave_fence();                           // WAR vs previous iteration's reads
#pragma unroll
    for (int j = 0; j < 8; ++j) {
      pl[(h * 8 + j) * 32 + r]      = (_Float16)s0[j];
      pl[(h * 8 + j) * 32 + 16 + r] = (_Float16)s1[j];
    }
    wave_fence();                           // RAW: stores before reads
    v16h pa;
    {
      union { v16h v; v8h p[2]; } u;
      u.p[0] = *(const v8h*)(pl + r * 32 + h * 16);
      u.p[1] = *(const v8h*)(pl + r * 32 + h * 16 + 8);
      pa = u.v;
    }

    // ---- O += P @ V from LDS (batched 4 loads + 4 WMMAs) ----
#pragma unroll
    for (int nt0 = 0; nt0 < 16; nt0 += 4) {
      v16h b0 = ld_bfrag(vlds + (nt0 * 16 + 0 * 16 + r) * 32, h);
      v16h b1 = ld_bfrag(vlds + (nt0 * 16 + 1 * 16 + r) * 32, h);
      v16h b2 = ld_bfrag(vlds + (nt0 * 16 + 2 * 16 + r) * 32, h);
      v16h b3 = ld_bfrag(vlds + (nt0 * 16 + 3 * 16 + r) * 32, h);
      acc[nt0 + 0] = wmma_f16(pa, b0, acc[nt0 + 0]);
      acc[nt0 + 1] = wmma_f16(pa, b1, acc[nt0 + 1]);
      acc[nt0 + 2] = wmma_f16(pa, b2, acc[nt0 + 2]);
      acc[nt0 + 3] = wmma_f16(pa, b3, acc[nt0 + 3]);
    }
  }

  // ---- normalize and emit O (f16, row-major) ----
#pragma unroll
  for (int nt = 0; nt < 16; ++nt) {
    int col = nt * 16 + r;
#pragma unroll
    for (int j = 0; j < 8; ++j) {
      int row = m0 + h * 8 + j;
      O[((size_t)(b * kS + row)) * kC + col] = (_Float16)(acc[nt][j] / lrow[j]);
    }
  }
}

// ---------------- Output projection + bias + residual ----------------------
__global__ __launch_bounds__(256, 1) void sa_oproj(const _Float16* __restrict__ O,
                                                   const _Float16* __restrict__ wpack,
                                                   const float* __restrict__ bo,
                                                   const float* __restrict__ x,
                                                   float* __restrict__ out) {
  const int lane = threadIdx.x & 31, wave = threadIdx.x >> 5;
  const int b = blockIdx.y;
  const int m0 = blockIdx.x * 128 + wave * 16;
  const int h = lane >> 4, r = lane & 15;
  const _Float16* wp = wpack + (size_t)3 * 65536;

  v8f acc[16];
#pragma unroll
  for (int nt = 0; nt < 16; ++nt) acc[nt] = vzero();

  const _Float16* arow = O + ((size_t)(b * kS + m0 + r)) * kC;
  v16h a = ld_afrag(arow, h);
#pragma unroll
  for (int kt = 0; kt < 8; ++kt) {
    v16h anext = a;
    if (kt < 7) anext = ld_afrag(arow + (kt + 1) * 32, h);
#pragma unroll
    for (int nt0 = 0; nt0 < 16; nt0 += 8) {
      v16h bb[8];
#pragma unroll
      for (int u = 0; u < 8; ++u)
        bb[u] = *(const v16h*)(wp + ((((nt0 + u) * 8 + kt) * 32 + lane) << 4));
#pragma unroll
      for (int u = 0; u < 8; ++u)
        acc[nt0 + u] = wmma_f16(a, bb[u], acc[nt0 + u]);
    }
    a = anext;
  }

#pragma unroll
  for (int nt = 0; nt < 16; ++nt) {
    int col = nt * 16 + r;
    float bval = bo[col];
#pragma unroll
    for (int j = 0; j < 8; ++j) {
      int row = m0 + h * 8 + j;
      size_t idx = ((size_t)(b * kS + row)) * kC + col;
      out[idx] = acc[nt][j] + bval + x[idx];
    }
  }
}

// ---------------------------------------------------------------------------
extern "C" void kernel_launch(void* const* d_in, const int* in_sizes, int n_in,
                              void* d_out, int out_size, void* d_ws, size_t ws_size,
                              hipStream_t stream) {
  (void)in_sizes; (void)n_in; (void)out_size; (void)ws_size;  // needs ~68 MB ws
  const float* x   = (const float*)d_in[0];
  const float* gsc = (const float*)d_in[1];
  const float* gbi = (const float*)d_in[2];
  const float* wq  = (const float*)d_in[3];
  const float* bq  = (const float*)d_in[4];
  const float* wk  = (const float*)d_in[5];
  const float* bk  = (const float*)d_in[6];
  const float* wv  = (const float*)d_in[7];
  const float* bv  = (const float*)d_in[8];
  const float* wo  = (const float*)d_in[9];
  const float* bo  = (const float*)d_in[10];
  float* out = (float*)d_out;

  char* ws = (char*)d_ws;
  float*     stats = (float*)(ws + 0);
  _Float16*  hf    = (_Float16*)(ws + OFF_HF);    // reused as O after attention
  _Float16*  Qb    = (_Float16*)(ws + OFF_Q);
  _Float16*  Kb    = (_Float16*)(ws + OFF_K);
  _Float16*  Vt    = (_Float16*)(ws + OFF_V);
  _Float16*  wpack = (_Float16*)(ws + OFF_W);

  sa_gn_stats<<<dim3(32, kB), 256, 0, stream>>>(x, stats);
  sa_gn_apply<<<dim3((kB * kS * kC / 8) / 256), 256, 0, stream>>>(x, gsc, gbi, stats, hf);
  sa_pack_w  <<<dim3(256, 4), 256, 0, stream>>>(wq, wk, wv, wo, wpack);
  sa_qkv     <<<dim3(kS / 128, kB, 3), 256, 0, stream>>>(hf, wpack, bq, bk, bv, Qb, Kb, Vt);
  sa_attn    <<<dim3(kS / 128, kB), 256, 0, stream>>>(Qb, Kb, Vt, hf);
  sa_oproj   <<<dim3(kS / 128, kB), 256, 0, stream>>>(hf, wpack, bo, x, out);
}
